// VisionModel_43078521979076
// MI455X (gfx1250) — matrix operand order repaired and checked by reading, not compiled, not run on hardware
//
#include <hip/hip_runtime.h>

typedef __attribute__((ext_vector_type(16))) _Float16 v16h;
typedef __attribute__((ext_vector_type(8)))  _Float16 v8h;
typedef __attribute__((ext_vector_type(8)))  float    v8f;

__device__ __forceinline__ v16h cat8(v8h lo, v8h hi) {
  return __builtin_shufflevector(lo, hi, 0,1,2,3,4,5,6,7,8,9,10,11,12,13,14,15);
}

// ---------------------------------------------------------------------------
// Weight pack: OIHW f32 -> [O][kpos][Ipad] f16 (Cin contiguous => B-fragment
// is one aligned 32B load per lane). Ipad pads Cin=3 -> 32 for layer 1.
// ---------------------------------------------------------------------------
__global__ void pack_weights_k(const float* __restrict__ w, _Float16* __restrict__ wp,
                               int O, int I, int Ipad, int KK) {
  int idx = blockIdx.x * blockDim.x + threadIdx.x;
  int total = O * KK * Ipad;
  if (idx >= total) return;
  int c = idx % Ipad;
  int t = idx / Ipad;
  int r = t % KK;
  int o = t / KK;
  float v = (c < I) ? w[((size_t)o * I + c) * KK + r] : 0.0f;
  wp[idx] = (_Float16)v;
}

// x: NCHW f32 (B,3,H,W) -> NHWC f16 (B,H,W,32) zero-padded channels
__global__ void pack_input_k(const float* __restrict__ x, _Float16* __restrict__ xp,
                             int Bn, int C, int Cpad, int HW) {
  int idx = blockIdx.x * blockDim.x + threadIdx.x;
  int total = Bn * HW * Cpad;
  if (idx >= total) return;
  int c = idx % Cpad;
  int p = (idx / Cpad) % HW;
  int b = idx / (Cpad * HW);
  float v = (c < C) ? x[((size_t)b * C + c) * HW + p] : 0.0f;
  xp[idx] = (_Float16)v;
}

// ---------------------------------------------------------------------------
// Implicit-GEMM conv via v_wmma_f32_16x16x32_f16 with LDS-staged activations.
//
// Block = 16 output pixels (one row strip) x COUT channels; wave w owns the
// 16-wide cout tile w*16. The (KS x (16+KS-1) x CIN) halo patch is staged
// once into LDS by all waves (vector v8h global loads, borders zero-filled,
// x2-nearest upsample folded into the source address), then every wave reads
// its A fragments via ds_load_b128 -- 9*CIN/32 reuses of a ~14KB patch
// instead of per-wave global gathers.
//
// Fragment maps (wave32, contiguous k order in A and B, not the ISA table order):
//  A (16x32 f16): lane&15 = M(pixel), elems 0..15 = K(half*16..+15)
//                 -> two adjacent v8h LDS loads.
//  B (32x16 f16): lane&15 = N(cout), elems 0..15 = K(half*16..+15)
//                 -> one aligned v16h global load from packed weights.
//  D (16x16 f32): elem r -> M = r + 8*half, N = lane&15.
// CSPLIT feeds concat([in0,in1],C) without materializing the concat.
// LDS pixel row padded by 8 halves so M-lanes land in distinct bank groups.
// ---------------------------------------------------------------------------
template<int CIN, int CSPLIT, int COUT, int KS, int UPS, int RELU, int OUTF32>
__global__ __launch_bounds__(32 * (COUT / 16))
void conv_wmma_k(const _Float16* __restrict__ in0, const _Float16* __restrict__ in1,
                 const _Float16* __restrict__ wp, const float* __restrict__ bias,
                 void* __restrict__ outv, int H, int W) {
  constexpr int KK      = KS * KS;
  constexpr int PADO    = (KS == 3) ? 1 : 0;
  constexpr int CIN2    = CIN - CSPLIT;
  constexpr int PATCH_H = KS;                 // 3 or 1
  constexpr int PATCH_W = 16 + 2 * PADO;      // 18 or 16
  constexpr int SROW    = CIN + 8;            // +16B pad: kills bank conflicts
  constexpr int NVEC    = PATCH_H * PATCH_W * (CIN / 8);

  __shared__ _Float16 smem[PATCH_H * PATCH_W * SROW];

  const int lane  = threadIdx.x & 31;
  const int wave  = threadIdx.x >> 5;
  const int half  = lane >> 4;
  const int mrow  = lane & 15;
  const int x0    = blockIdx.x << 4;
  const int y     = blockIdx.y;
  const int b     = blockIdx.z;
  const int cout0 = wave << 4;
  const int Hs = H >> UPS, Ws = W >> UPS;

  // ---- stage halo patch into LDS (8 channels / 16B per iteration) --------
  for (int i = threadIdx.x; i < NVEC; i += 32 * (COUT / 16)) {
    const int cv = i % (CIN / 8);
    const int pxp = (i / (CIN / 8)) % PATCH_W;
    const int ryp = i / ((CIN / 8) * PATCH_W);
    const int yy = y + ryp - PADO;            // post-upsample coords
    const int xx = x0 + pxp - PADO;
    v8h v = {};
    if (yy >= 0 && yy < H && xx >= 0 && xx < W) {
      const int ys = yy >> UPS;               // fold upsample into source
      const int xs = xx >> UPS;
      const long long pix = (long long)(b * Hs + ys) * Ws + xs;
      const int c = cv * 8;
      const _Float16* p = (c < CSPLIT) ? (in0 + pix * CSPLIT + c)
                                       : (in1 + pix * CIN2 + (c - CSPLIT));
      v = *(const v8h*)p;
    }
    *(v8h*)(smem + (size_t)(ryp * PATCH_W + pxp) * SROW + cv * 8) = v;
  }
  __syncthreads();

  // ---- K loop: 9 * CIN/32 WMMAs, A from LDS, B from L2-resident weights --
  v8f acc = {};
  #pragma unroll
  for (int r = 0; r < KK; ++r) {
    const int ky = r / KS;
    const int kx = r % KS;
    const _Float16* arow = smem + (size_t)(ky * PATCH_W + mrow + kx) * SROW;
    #pragma unroll
    for (int c0 = 0; c0 < CIN; c0 += 32) {
      const int cA = c0 + half * 16;          // elems 0..15 = K(half*16..+15)
      const v8h alo = *(const v8h*)(arow + cA);
      const v8h ahi = *(const v8h*)(arow + cA + 8);
      const v16h a = cat8(alo, ahi);
      const int ck = c0 + half * 16;          // B: 16 contiguous channels
      const v16h bm = *(const v16h*)(wp + ((size_t)(cout0 + mrow) * KK + r) * CIN + ck);
      acc = __builtin_amdgcn_wmma_f32_16x16x32_f16(false, a, false, bm,
                                                   (short)0, acc, false, false);
    }
  }

  const float bv = bias[cout0 + mrow];
  #pragma unroll
  for (int rr = 0; rr < 8; ++rr) {
    float v = acc[rr] + bv;
    if (RELU) v = v > 0.0f ? v : 0.0f;
    const int xo = x0 + rr + 8 * half;        // D: M = rr + 8*half
    if (OUTF32) {  // final feature map -> f32 NCHW (feeds cumsum)
      ((float*)outv)[(((size_t)b * COUT + cout0 + mrow) * H + y) * W + xo] = v;
    } else {       // f16 NHWC
      ((_Float16*)outv)[((size_t)(b * H + y) * W + xo) * COUT + cout0 + mrow] = (_Float16)v;
    }
  }
}

// 2x2 max pool, NHWC f16
__global__ void maxpool_k(const _Float16* __restrict__ in, _Float16* __restrict__ out,
                          int Bn, int Ho, int Wo, int C) {
  int idx = blockIdx.x * blockDim.x + threadIdx.x;
  int total = Bn * Ho * Wo * C;
  if (idx >= total) return;
  int c = idx % C;
  int x = (idx / C) % Wo;
  int y = (idx / (C * Wo)) % Ho;
  int b = idx / (C * Wo * Ho);
  const int Wi = Wo * 2;
  size_t base = ((size_t)(b * 2 * Ho + 2 * y) * Wi + 2 * x) * C + c;
  float v0 = (float)in[base];
  float v1 = (float)in[base + C];
  float v2 = (float)in[base + (size_t)Wi * C];
  float v3 = (float)in[base + (size_t)Wi * C + C];
  out[idx] = (_Float16)fmaxf(fmaxf(v0, v1), fmaxf(v2, v3));
}

// integral image pass 1: per (b,c,y) row prefix along x. ii is (B,C,H+1,W+1).
__global__ void cumsum_x_k(const float* __restrict__ f, float* __restrict__ ii,
                           int rows, int H, int W) {
  int row = blockIdx.x * blockDim.x + threadIdx.x;
  if (row >= rows) return;
  int y  = row % H;
  int bc = row / H;
  const float* src = f + (size_t)row * W;
  float* d = ii + ((size_t)bc * (H + 1) + (y + 1)) * (W + 1);
  d[0] = 0.0f;
  float s = 0.0f;
  for (int x = 0; x < W; ++x) { s += src[x]; d[x + 1] = s; }
}

// integral image pass 2: per (b,c,x) column prefix along y (+ zero top row).
__global__ void cumsum_y_k(float* __restrict__ ii, int cols, int H, int W) {
  int col = blockIdx.x * blockDim.x + threadIdx.x;
  if (col >= cols) return;
  int x  = col % (W + 1);
  int bc = col / (W + 1);
  float* d = ii + (size_t)bc * (H + 1) * (W + 1) + x;
  d[0] = 0.0f;
  float s = 0.0f;
  for (int y = 1; y <= H; ++y) {
    s += d[(size_t)y * (W + 1)];
    d[(size_t)y * (W + 1)] = s;
  }
}

// adaptive 7x7 pool via 4 integral-image corner reads.
// pooled layout (B, C + R*C, 49): n<C -> global box, else n-C = roi*C + c.
__global__ void roi_pool_k(const float* __restrict__ ii, const int* __restrict__ rois,
                           float* __restrict__ pooled, int Bn, int C, int R, int H, int W) {
  const int NCH = C + R * C;
  int idx = blockIdx.x * blockDim.x + threadIdx.x;
  int total = Bn * NCH * 49;
  if (idx >= total) return;
  int k = idx % 49;
  int n = (idx / 49) % NCH;
  int b = idx / (49 * NCH);
  int ox = k % 7, oy = k / 7;
  int x1, y1, x2, y2, c;
  if (n < C) { c = n; x1 = 0; y1 = 0; x2 = W; y2 = H; }
  else {
    int j = n - C; int roi = j / C; c = j % C;
    x1 = rois[roi * 4 + 0]; y1 = rois[roi * 4 + 1];
    x2 = rois[roi * 4 + 2]; y2 = rois[roi * 4 + 3];
  }
  int h = y2 - y1, w = x2 - x1;
  int sy = y1 + (oy * h) / 7;
  int ey = y1 + ((oy + 1) * h + 6) / 7;
  int sx = x1 + (ox * w) / 7;
  int ex = x1 + ((ox + 1) * w + 6) / 7;
  const float* p = ii + (size_t)(b * C + c) * (H + 1) * (W + 1);
  float s = p[(size_t)ey * (W + 1) + ex] - p[(size_t)sy * (W + 1) + ex]
          - p[(size_t)ey * (W + 1) + sx] + p[(size_t)sy * (W + 1) + sx];
  float area = (float)((ey - sy) * (ex - sx));
  pooled[idx] = s / area;
}

// final FC: out[b,n,f] = sum_k pooled[b,n,k]*fcw[f,k] + fcb[f]  (K=49, tiny)
__global__ void fc_k(const float* __restrict__ pooled, const float* __restrict__ fcw,
                     const float* __restrict__ fcb, float* __restrict__ out,
                     int BN, int F) {
  int idx = blockIdx.x * blockDim.x + threadIdx.x;
  if (idx >= BN * F) return;
  int f = idx % F;
  int r = idx / F;
  const float* pr = pooled + (size_t)r * 49;
  const float* wr = fcw + (size_t)f * 49;
  float s = fcb[f];
  #pragma unroll
  for (int k = 0; k < 49; ++k) s += pr[k] * wr[k];
  out[idx] = s;
}

// ---------------------------------------------------------------------------
extern "C" void kernel_launch(void* const* d_in, const int* in_sizes, int n_in,
                              void* d_out, int out_size, void* d_ws, size_t ws_size,
                              hipStream_t stream) {
  (void)in_sizes; (void)n_in; (void)out_size; (void)ws_size;
  const int B = 8, IMG = 256, R = 32, FOUT = 512;

  const float* x    = (const float*)d_in[0];
  const int*   rois = (const int*)d_in[1];
  const float* e1w1 = (const float*)d_in[2];  const float* e1b1 = (const float*)d_in[3];
  const float* e1w2 = (const float*)d_in[4];  const float* e1b2 = (const float*)d_in[5];
  const float* e2w1 = (const float*)d_in[6];  const float* e2b1 = (const float*)d_in[7];
  const float* e2w2 = (const float*)d_in[8];  const float* e2b2 = (const float*)d_in[9];
  const float* bw1  = (const float*)d_in[10]; const float* bb1  = (const float*)d_in[11];
  const float* bw2  = (const float*)d_in[12]; const float* bb2  = (const float*)d_in[13];
  const float* u2w  = (const float*)d_in[14]; const float* u2b  = (const float*)d_in[15];
  const float* d2w1 = (const float*)d_in[16]; const float* d2b1 = (const float*)d_in[17];
  const float* d2w2 = (const float*)d_in[18]; const float* d2b2 = (const float*)d_in[19];
  const float* u1w  = (const float*)d_in[20]; const float* u1b  = (const float*)d_in[21];
  const float* d1w1 = (const float*)d_in[22]; const float* d1b1 = (const float*)d_in[23];
  const float* d1w2 = (const float*)d_in[24]; const float* d1b2 = (const float*)d_in[25];
  const float* fw   = (const float*)d_in[26]; const float* fb   = (const float*)d_in[27];
  const float* fcw  = (const float*)d_in[28]; const float* fcb  = (const float*)d_in[29];

  // ---- workspace bump allocator with lifetime-based aliasing -------------
  char* base = (char*)d_ws;
  size_t off = 0;
  auto take = [&](size_t bytes) -> void* {
    void* p = base + off; off += (bytes + 255) & ~(size_t)255; return p;
  };
  // packed f16 weights (persist, ~1 MB, L2-resident)
  _Float16* wpe1w1 = (_Float16*)take((size_t)32 * 9 * 32 * 2);
  _Float16* wpe1w2 = (_Float16*)take((size_t)32 * 9 * 32 * 2);
  _Float16* wpe2w1 = (_Float16*)take((size_t)64 * 9 * 32 * 2);
  _Float16* wpe2w2 = (_Float16*)take((size_t)64 * 9 * 64 * 2);
  _Float16* wpbw1  = (_Float16*)take((size_t)128 * 9 * 64 * 2);
  _Float16* wpbw2  = (_Float16*)take((size_t)128 * 9 * 128 * 2);
  _Float16* wpu2w  = (_Float16*)take((size_t)64 * 9 * 128 * 2);
  _Float16* wpd2w1 = (_Float16*)take((size_t)64 * 9 * 128 * 2);
  _Float16* wpd2w2 = (_Float16*)take((size_t)64 * 9 * 64 * 2);
  _Float16* wpu1w  = (_Float16*)take((size_t)32 * 9 * 64 * 2);
  _Float16* wpd1w1 = (_Float16*)take((size_t)32 * 9 * 64 * 2);
  _Float16* wpd1w2 = (_Float16*)take((size_t)32 * 9 * 32 * 2);
  _Float16* wpfw   = (_Float16*)take((size_t)32 * 1 * 32 * 2);
  // persistent skip activations
  _Float16* E1 = (_Float16*)take(33554432);   // 8*256*256*32 f16
  _Float16* E2 = (_Float16*)take(16777216);   // 8*128*128*64 f16
  // transient slabs (aliased across the layer chain)
  char* T1 = (char*)take(67108864);
  char* T2 = (char*)take(33554432);
  char* T3 = (char*)take(33554432);
  float* II = (float*)take((size_t)8 * 32 * 257 * 257 * 4);

  auto cdiv = [](int a, int b) { return (a + b - 1) / b; };

  // ---- pack weights + input ---------------------------------------------
  pack_weights_k<<<cdiv(32*9*32,256),256,0,stream>>>(e1w1, wpe1w1, 32, 3, 32, 9);
  pack_weights_k<<<cdiv(32*9*32,256),256,0,stream>>>(e1w2, wpe1w2, 32, 32, 32, 9);
  pack_weights_k<<<cdiv(64*9*32,256),256,0,stream>>>(e2w1, wpe2w1, 64, 32, 32, 9);
  pack_weights_k<<<cdiv(64*9*64,256),256,0,stream>>>(e2w2, wpe2w2, 64, 64, 64, 9);
  pack_weights_k<<<cdiv(128*9*64,256),256,0,stream>>>(bw1, wpbw1, 128, 64, 64, 9);
  pack_weights_k<<<cdiv(128*9*128,256),256,0,stream>>>(bw2, wpbw2, 128, 128, 128, 9);
  pack_weights_k<<<cdiv(64*9*128,256),256,0,stream>>>(u2w, wpu2w, 64, 128, 128, 9);
  pack_weights_k<<<cdiv(64*9*128,256),256,0,stream>>>(d2w1, wpd2w1, 64, 128, 128, 9);
  pack_weights_k<<<cdiv(64*9*64,256),256,0,stream>>>(d2w2, wpd2w2, 64, 64, 64, 9);
  pack_weights_k<<<cdiv(32*9*64,256),256,0,stream>>>(u1w, wpu1w, 32, 64, 64, 9);
  pack_weights_k<<<cdiv(32*9*64,256),256,0,stream>>>(d1w1, wpd1w1, 32, 64, 64, 9);
  pack_weights_k<<<cdiv(32*9*32,256),256,0,stream>>>(d1w2, wpd1w2, 32, 32, 32, 9);
  pack_weights_k<<<cdiv(32*1*32,256),256,0,stream>>>(fw, wpfw, 32, 32, 32, 1);

  _Float16* XP = (_Float16*)T2;
  pack_input_k<<<cdiv(B*IMG*IMG*32,256),256,0,stream>>>(x, XP, B, 3, 32, IMG*IMG);

  // ---- U-Net (all WMMA convs, f16 NHWC activations) ----------------------
  _Float16* E1A = (_Float16*)T1;
  conv_wmma_k<32,32,32,3,0,1,0><<<dim3(16,256,B), 64, 0, stream>>>(XP, XP, wpe1w1, e1b1, E1A, 256, 256);
  conv_wmma_k<32,32,32,3,0,1,0><<<dim3(16,256,B), 64, 0, stream>>>(E1A, E1A, wpe1w2, e1b2, E1, 256, 256);

  _Float16* PL1 = (_Float16*)T2;
  maxpool_k<<<cdiv(B*128*128*32,256),256,0,stream>>>(E1, PL1, B, 128, 128, 32);

  _Float16* E2A = (_Float16*)T1;
  conv_wmma_k<32,32,64,3,0,1,0><<<dim3(8,128,B), 128, 0, stream>>>(PL1, PL1, wpe2w1, e2b1, E2A, 128, 128);
  conv_wmma_k<64,64,64,3,0,1,0><<<dim3(8,128,B), 128, 0, stream>>>(E2A, E2A, wpe2w2, e2b2, E2, 128, 128);

  _Float16* PL2 = (_Float16*)T3;
  maxpool_k<<<cdiv(B*64*64*64,256),256,0,stream>>>(E2, PL2, B, 64, 64, 64);

  _Float16* BA = (_Float16*)T1;
  conv_wmma_k<64,64,128,3,0,1,0><<<dim3(4,64,B), 256, 0, stream>>>(PL2, PL2, wpbw1, bb1, BA, 64, 64);
  _Float16* BO = (_Float16*)T2;
  conv_wmma_k<128,128,128,3,0,1,0><<<dim3(4,64,B), 256, 0, stream>>>(BA, BA, wpbw2, bb2, BO, 64, 64);

  _Float16* U2 = (_Float16*)T1;   // upsample folded into conv (UPS=1)
  conv_wmma_k<128,128,64,3,1,1,0><<<dim3(8,128,B), 128, 0, stream>>>(BO, BO, wpu2w, u2b, U2, 128, 128);

  _Float16* D2A = (_Float16*)T3;  // concat(U2, E2) via split channels
  conv_wmma_k<128,64,64,3,0,1,0><<<dim3(8,128,B), 128, 0, stream>>>(U2, E2, wpd2w1, d2b1, D2A, 128, 128);
  _Float16* D2 = (_Float16*)T2;
  conv_wmma_k<64,64,64,3,0,1,0><<<dim3(8,128,B), 128, 0, stream>>>(D2A, D2A, wpd2w2, d2b2, D2, 128, 128);

  _Float16* U1 = (_Float16*)T1;
  conv_wmma_k<64,64,32,3,1,1,0><<<dim3(16,256,B), 64, 0, stream>>>(D2, D2, wpu1w, u1b, U1, 256, 256);

  _Float16* D1A = (_Float16*)T3;  // concat(U1, E1)
  conv_wmma_k<64,32,32,3,0,1,0><<<dim3(16,256,B), 64, 0, stream>>>(U1, E1, wpd1w1, d1b1, D1A, 256, 256);
  _Float16* D1 = (_Float16*)T2;
  conv_wmma_k<32,32,32,3,0,1,0><<<dim3(16,256,B), 64, 0, stream>>>(D1A, D1A, wpd1w2, d1b2, D1, 256, 256);

  // final 1x1 conv, no relu, f32 NCHW output (feeds integral image)
  float* F = (float*)T1;
  conv_wmma_k<32,32,32,1,0,0,1><<<dim3(16,256,B), 64, 0, stream>>>(D1, D1, wpfw, fb, F, 256, 256);

  // ---- integral image (f32 for range), ROI pool, FC ----------------------
  cumsum_x_k<<<cdiv(B*32*256,256),256,0,stream>>>(F, II, B*32*256, 256, 256);
  cumsum_y_k<<<cdiv(B*32*257,256),256,0,stream>>>(II, B*32*257, 256, 256);

  float* pooled = (float*)T3;     // (B, 32 + R*32, 49)
  const int NCH = 32 + R * 32;
  roi_pool_k<<<cdiv(B*NCH*49,256),256,0,stream>>>(II, rois, pooled, B, 32, R, 256, 256);

  fc_k<<<cdiv(B*NCH*FOUT,256),256,0,stream>>>(pooled, fcw, fcb, (float*)d_out, B*NCH, FOUT);
}
